// GroupConvLayer_962072674967
// MI455X (gfx1250) — compile-verified, matrix operand order checked by reading, not run on hardware
//
#include <hip/hip_runtime.h>

typedef __bf16 bf16;
typedef __attribute__((ext_vector_type(16))) __bf16 v16bf;
typedef __attribute__((ext_vector_type(8)))  float  v8f;
typedef unsigned int v4u __attribute__((ext_vector_type(4)));
typedef int          v8i __attribute__((ext_vector_type(8)));
typedef int          v4i __attribute__((ext_vector_type(4)));

#define BATCH 2048
#define GSZ   24
#define CIN   512
#define COUT  512

// Tile geometry: block = 128(M) x 64(N), K-step = 64. 8 waves, each 32x32.
#define BM 128
#define BN 64
// LDS layout written by TDM: every 64B (32 halfs) chunk followed by 16B pad
// => chunk stride 80B, row (64 halfs = 2 chunks) stride 160B.
#define CHB   80
#define ROWB  160
#define ABYTES (BM * ROWB)            // 20480
#define BBYTES (BN * ROWB)            // 10240
#define BUFB   (ABYTES + BBYTES)      // 30720 per buffer, x2 = 61440

#if defined(__has_builtin)
#if __has_builtin(__builtin_amdgcn_tensor_load_to_lds)
#define USE_TDM 1
#endif
#endif

// ---------------- pack kernels ----------------

__global__ void pack_x_kernel(const float* __restrict__ x, bf16* __restrict__ xp, size_t n) {
  size_t idx = (size_t)blockIdx.x * blockDim.x + threadIdx.x;
  if (idx < n) xp[idx] = (bf16)x[idx];
}

// wp[orb][o][c] = w[o][c][orb]   (w is (C_OUT, C_IN, 24) f32)
__global__ void pack_w_kernel(const float* __restrict__ w, bf16* __restrict__ wp) {
  size_t idx = (size_t)blockIdx.x * blockDim.x + threadIdx.x;
  const size_t n = (size_t)GSZ * COUT * CIN;
  if (idx >= n) return;
  int orb = (int)(idx / ((size_t)COUT * CIN));
  int rem = (int)(idx % ((size_t)COUT * CIN));
  int o = rem / CIN;
  int c = rem % CIN;
  wp[idx] = (bf16)w[((size_t)o * CIN + c) * GSZ + orb];
}

// ---------------- TDM: 2D tile -> LDS, one instruction ----------------
#ifdef USE_TDM
// D# packing per CDNA5 ISA 8.3/8.4: group0 = {flags, lds_addr, gaddr_lo, gaddr_hi|type},
// group1 = {pad ctrl, tensor_dim0/1, tile_dim0/1/2, strides}. 2D tensor, width 64 elems,
// data_size=2B, pad 4 DWORDs every 16 DWORDs (64B -> stride 80B in LDS).
__device__ __forceinline__ void tdm_load_2d(unsigned lds_addr, const void* gp,
                                            unsigned rows, unsigned stride_elems) {
  const unsigned long long ga = (unsigned long long)(size_t)gp;
  v4u g0 = { 1u,                                   // count=1, user mode
             lds_addr,                             // lds_addr (bytes)
             (unsigned)ga,                         // global_addr[31:0]
             (unsigned)(ga >> 32) | (2u << 30) };  // global_addr[56:32] | type=2
  v8i g1;
  g1[0] = (1 << 16)                    // data_size = 2 bytes
        | (1 << 20)                    // pad_enable
        | (3 << 22)                    // pad_interval: 16 DWORDs (64B)
        | (3 << 25);                   // pad_amount: 4 DWORDs (16B)
  g1[1] = (int)(64u << 16);            // tensor_dim0 = 64 (low 16 bits)
  g1[2] = (int)(rows << 16);           // tensor_dim0 hi=0 | tensor_dim1 lo16 = rows
  g1[3] = (int)(64u << 16);            // tensor_dim1 hi=0 | tile_dim0 = 64
  g1[4] = (int)rows;                   // tile_dim1 = rows | tile_dim2 = 0
  g1[5] = (int)stride_elems;           // tensor_dim0_stride lo32
  g1[6] = 0;                           // stride hi | tensor_dim1_stride lo
  g1[7] = 0;
  v4i zz4 = { 0, 0, 0, 0 };
  v8i zz8 = { 0, 0, 0, 0, 0, 0, 0, 0 };
  // 6-arg form (this toolchain): (g0, g1, g2, g3, g4, cpol)
  __builtin_amdgcn_tensor_load_to_lds(g0, g1, zz4, zz4, zz8, 0);
}

__device__ __forceinline__ void tdm_wait0() {
#if __has_builtin(__builtin_amdgcn_s_wait_tensorcnt)
  __builtin_amdgcn_s_wait_tensorcnt(0);
#else
  asm volatile("s_wait_tensorcnt 0x0" ::: "memory");
#endif
}
#endif

// ---------------- fragment loader ----------------
// 16-bit A/B layout (ISA 7.12.2): lane holds one row (A: M, B col-major: N);
// lanes 0-15: K = {0..7} u {16..23}; lanes 16-31: K = {8..15} u {24..31}.
// `p` points at a 32-half chunk stored contiguously (64B) in LDS.
__device__ __forceinline__ v16bf load_frag(const char* p, int kb) {
  union { v16bf v; unsigned u[8]; } f;
  const unsigned* p0 = (const unsigned*)(p + kb * 2);
  const unsigned* p1 = (const unsigned*)(p + kb * 2 + 32);
  f.u[0] = p0[0]; f.u[1] = p0[1]; f.u[2] = p0[2]; f.u[3] = p0[3];
  f.u[4] = p1[0]; f.u[5] = p1[1]; f.u[6] = p1[2]; f.u[7] = p1[3];
  return f.v;
}

#ifndef USE_TDM
// Fallback staging: per-lane b128 global loads + ds stores into the same layout.
__device__ __forceinline__ void stage_tiles(const bf16* gA, const bf16* gB,
                                            char* Abuf, char* Bbuf, int t) {
  #pragma unroll
  for (int rep = 0; rep < 4; ++rep) {             // A: 128 rows x 8 segs
    int idx = rep * 256 + t;
    int row = idx >> 3, seg = idx & 7;
    uint4 v = *(const uint4*)(gA + (size_t)row * (GSZ * CIN) + seg * 8);
    *(uint4*)(Abuf + row * ROWB + (seg >> 2) * CHB + (seg & 3) * 16) = v;
  }
  #pragma unroll
  for (int rep = 0; rep < 2; ++rep) {             // B: 64 rows x 8 segs
    int idx = rep * 256 + t;
    int row = idx >> 3, seg = idx & 7;
    uint4 v = *(const uint4*)(gB + (size_t)row * CIN + seg * 8);
    *(uint4*)(Bbuf + row * ROWB + (seg >> 2) * CHB + (seg & 3) * 16) = v;
  }
}
#endif

// ---------------- main GEMM ----------------
// out[bBase:+128, i, oBase:+64] = sum_{j,c} x[b,j,c] * Wp[orb(i,j)][o][c]
__global__ __launch_bounds__(256, 2)
void gemm_kernel(const bf16* __restrict__ Xp, const bf16* __restrict__ Wp,
                 const float* __restrict__ bias, const int* __restrict__ pair_orbit,
                 float* __restrict__ out) {
  __shared__ char smem[2 * BUFB];

  const int oBase = blockIdx.x * BN;
  const int bBase = blockIdx.y * BM;
  const int i     = blockIdx.z;

  const int t    = threadIdx.x;
  const int lane = t & 31;
  const int wave = t >> 5;        // 0..7
  const int wm   = wave & 3;      // 4 row-blocks of 32
  const int wn   = wave >> 2;     // 2 col-blocks of 32
  const int nl   = lane & 15;
  const int kb   = (lane < 16) ? 0 : 8;

  v8f accs[2][2] = {};

  const int TOTAL = GSZ * (CIN / 64);   // 24 j x 8 k-chunks = 192 steps

#ifdef USE_TDM
  const unsigned smemOff = (unsigned)(size_t)&smem[0];  // generic LDS addr low 32 = LDS offset
  int cur = 0;
  if (wave == 0) {
    const int orb0 = pair_orbit[i * GSZ + 0];
    tdm_load_2d(smemOff,          Xp + ((size_t)bBase * GSZ) * CIN,          BM, GSZ * CIN);
    tdm_load_2d(smemOff + ABYTES, Wp + ((size_t)orb0 * COUT + oBase) * CIN,  BN, CIN);
  }
  for (int s = 0; s < TOTAL; ++s) {
    if (wave == 0) tdm_wait0();   // current buffer's DMA complete (wave 0 issued it)
    __syncthreads();              // ...now visible to every wave

    if (wave == 0 && s + 1 < TOTAL) {   // issue next tile into the other buffer
      const int jn = (s + 1) >> 3, ckn = (s + 1) & 7;
      const int orb = pair_orbit[i * GSZ + jn];
      const unsigned dst = smemOff + (cur ^ 1) * BUFB;
      tdm_load_2d(dst,          Xp + (((size_t)bBase * GSZ + jn) * CIN) + ckn * 64, BM, GSZ * CIN);
      tdm_load_2d(dst + ABYTES, Wp + (((size_t)orb * COUT + oBase) * CIN) + ckn * 64, BN, CIN);
    }

    const char* Abuf = smem + cur * BUFB;
    const char* Bbuf = Abuf + ABYTES;
    #pragma unroll
    for (int q = 0; q < 2; ++q) {       // two 32-K chunks per step
      v16bf a[2], b[2];
      a[0] = load_frag(Abuf + (wm * 32 +      nl) * ROWB + q * CHB, kb);
      a[1] = load_frag(Abuf + (wm * 32 + 16 + nl) * ROWB + q * CHB, kb);
      b[0] = load_frag(Bbuf + (wn * 32 +      nl) * ROWB + q * CHB, kb);
      b[1] = load_frag(Bbuf + (wn * 32 + 16 + nl) * ROWB + q * CHB, kb);
      #pragma unroll
      for (int mi = 0; mi < 2; ++mi)
        #pragma unroll
        for (int ni = 0; ni < 2; ++ni)
          accs[mi][ni] = __builtin_amdgcn_wmma_f32_16x16x32_bf16(
              false, a[mi], false, b[ni], (short)0, accs[mi][ni], false, false);
    }
    __syncthreads();              // all waves done reading `cur` before it is refilled
    cur ^= 1;
  }
#else
  for (int s = 0; s < TOTAL; ++s) {
    const int j = s >> 3, ck = s & 7;
    const int orb = pair_orbit[i * GSZ + j];
    const bf16* gA = Xp + (((size_t)bBase * GSZ + j) * CIN) + ck * 64;
    const bf16* gB = Wp + (((size_t)orb * COUT + oBase) * CIN) + ck * 64;
    __syncthreads();
    stage_tiles(gA, gB, smem, smem + ABYTES, t);
    __syncthreads();
    #pragma unroll
    for (int q = 0; q < 2; ++q) {
      v16bf a[2], b[2];
      a[0] = load_frag(smem + (wm * 32 +      nl) * ROWB + q * CHB, kb);
      a[1] = load_frag(smem + (wm * 32 + 16 + nl) * ROWB + q * CHB, kb);
      b[0] = load_frag(smem + ABYTES + (wn * 32 +      nl) * ROWB + q * CHB, kb);
      b[1] = load_frag(smem + ABYTES + (wn * 32 + 16 + nl) * ROWB + q * CHB, kb);
      #pragma unroll
      for (int mi = 0; mi < 2; ++mi)
        #pragma unroll
        for (int ni = 0; ni < 2; ++ni)
          accs[mi][ni] = __builtin_amdgcn_wmma_f32_16x16x32_bf16(
              false, a[mi], false, b[ni], (short)0, accs[mi][ni], false, false);
    }
  }
#endif

  // Epilogue. C layout: VGPR r -> lanes 0-15: M=r; lanes 16-31: M=r+8; N=lane%16.
  const int m8 = (lane >> 4) << 3;
  const int n  = lane & 15;
  #pragma unroll
  for (int mi = 0; mi < 2; ++mi) {
    const int rb = bBase + wm * 32 + mi * 16 + m8;
    #pragma unroll
    for (int ni = 0; ni < 2; ++ni) {
      const int oc = oBase + wn * 32 + ni * 16 + n;
      const float bv = bias[oc];
      #pragma unroll
      for (int r = 0; r < 8; ++r)
        out[((size_t)(rb + r) * GSZ + i) * COUT + oc] = accs[mi][ni][r] + bv;
    }
  }
}

// ---------------- host launcher ----------------

extern "C" void kernel_launch(void* const* d_in, const int* in_sizes, int n_in,
                              void* d_out, int out_size, void* d_ws, size_t ws_size,
                              hipStream_t stream) {
  const float* x          = (const float*)d_in[0];   // (2048, 24, 512) f32
  const float* w          = (const float*)d_in[1];   // (512, 512, 24) f32
  const float* bias       = (const float*)d_in[2];   // (512,) f32
  const int*   pair_orbit = (const int*)d_in[3];     // (24, 24) i32
  float* out = (float*)d_out;                        // (2048, 24, 512) f32

  const size_t nx = (size_t)BATCH * GSZ * CIN;
  const size_t nw = (size_t)GSZ * COUT * CIN;

  bf16* Xp = (bf16*)d_ws;
  bf16* Wp = Xp + nx;

  pack_x_kernel<<<(unsigned)((nx + 255) / 256), 256, 0, stream>>>(x, Xp, nx);
  pack_w_kernel<<<(unsigned)((nw + 255) / 256), 256, 0, stream>>>(w, Wp);

  dim3 grid(COUT / BN, BATCH / BM, GSZ);   // (8, 16, 24)
  gemm_kernel<<<grid, 256, 0, stream>>>(Xp, Wp, bias, pair_orbit, out);
}
